// NeuralControlCritic_50646254354837
// MI455X (gfx1250) — matrix operand order repaired and checked by reading, not compiled, not run on hardware
//
#include <hip/hip_runtime.h>
#include <hip/hip_bf16.h>

// ---------------------------------------------------------------------------
// NeuralControlCritic on gfx1250 (MI455X):
//   h1 = relu(x @ w_in1 + b)        [256x1024]@[1024x8192]
//   h2 = h1 @ w_in2 + b             [256x8192]@[8192x8192]
//   2x Mamba2 blocks (in_proj GEMM, conv+silu, softplus dt, SSM scan,
//                     gated RMSNorm, out_proj GEMM)
//   out = relu(h @ w_out1 + b) @ w_out2 + b2
// GEMMs: fp32 weights read once from HBM, converted to bf16 in-register while
// staging to LDS, consumed by v_wmma_f32_16x16x32_bf16 with fp32 accumulate.
// Double-buffered LDS tiles; next tile's global loads issue before the WMMAs
// so HBM latency hides under matrix math.
// ---------------------------------------------------------------------------

typedef __attribute__((ext_vector_type(16))) __bf16 bf16x16;
typedef __attribute__((ext_vector_type(8)))  __bf16 bf16x8;
typedef __attribute__((ext_vector_type(8)))  float  f32x8;

#define SEQL     8
#define HID      1024
#define BATCH    256
#define D_INNER  2048
#define HEADDIM  64
#define NHEADS   32
#define DSTATE   128
#define DCONV    4
#define CONV_DIM 2304            // D_INNER + 2*DSTATE
#define DPROJ    4384            // 2*D_INNER + 2*DSTATE + NHEADS
#define ROWS     (BATCH * SEQL)  // 2048

#define BM  64
#define BN  64
#define BKT 64                   // K per stage (2 WMMA K-steps of 32)

// ------------------------- WMMA bf16 GEMM ----------------------------------
// C[M,N] = act(A[M,K] @ B[K,N] + bias).  A,B,C fp32 row-major in HBM.
// M % 64 == 0, K % 64 == 0 at all call sites; N edge guarded (block-uniform).
// 128 threads = 4 waves; wave w computes rows [w*16, w*16+16) of the 64x64
// block tile as 4 accumulators of 16x16.  8 WMMAs per stage per wave.
__global__ __launch_bounds__(128) void gemm_bf16_wmma(
    const float* __restrict__ A, const float* __restrict__ B,
    const float* __restrict__ bias, float* __restrict__ C,
    int M, int N, int K, int relu)
{
    __shared__ alignas(16) __bf16 As[2][BM][BKT];   // row-major  (M x K)
    __shared__ alignas(16) __bf16 Bt[2][BN][BKT];   // transposed (N x K)

    const int t    = threadIdx.x;
    const int wave = t >> 5;
    const int lane = t & 31;
    const int lo   = lane & 15;   // N (B/C/D) or M (A) index within tile
    const int hi   = lane >> 4;   // half-select per ISA 7.12.2 layouts
    const int m0   = blockIdx.y * BM;
    const int n0   = blockIdx.x * BN;
    const bool fastN = (n0 + BN <= N);     // block-uniform edge predicate

    // per-thread index decomposition shared by A and B staging (1024 float4s
    // per tile / 128 threads = 8 each)
    const int rA[8] = { (t+0*128)>>4, (t+1*128)>>4, (t+2*128)>>4, (t+3*128)>>4,
                        (t+4*128)>>4, (t+5*128)>>4, (t+6*128)>>4, (t+7*128)>>4 };
    const int cA    = (t & 15) << 2;       // column group 0..60 (same each iter)

    f32x8 acc[4];
    #pragma unroll
    for (int i = 0; i < 4; ++i)
        #pragma unroll
        for (int r = 0; r < 8; ++r) acc[i][r] = 0.0f;

    float4 ra[8], rb[8];

    // ---- issue global loads for one K-tile into registers ----------------
    auto loadRegs = [&](int kb) {
        #pragma unroll
        for (int i = 0; i < 8; ++i)
            ra[i] = *(const float4*)(A + (size_t)(m0 + rA[i]) * K + kb + cA);
        if (fastN) {
            #pragma unroll
            for (int i = 0; i < 8; ++i)
                rb[i] = *(const float4*)(B + (size_t)(kb + rA[i]) * N + n0 + cA);
        } else {
            #pragma unroll
            for (int i = 0; i < 8; ++i) {
                const int nb = n0 + cA;
                const float* src = B + (size_t)(kb + rA[i]) * N + nb;
                rb[i].x = (nb + 0 < N) ? src[0] : 0.0f;
                rb[i].y = (nb + 1 < N) ? src[1] : 0.0f;
                rb[i].z = (nb + 2 < N) ? src[2] : 0.0f;
                rb[i].w = (nb + 3 < N) ? src[3] : 0.0f;
            }
        }
    };

    // ---- convert fp32 -> bf16 and store registers into LDS buffer --------
    auto storeLDS = [&](int buf) {
        #pragma unroll
        for (int i = 0; i < 8; ++i) {
            As[buf][rA[i]][cA + 0] = (__bf16)ra[i].x;
            As[buf][rA[i]][cA + 1] = (__bf16)ra[i].y;
            As[buf][rA[i]][cA + 2] = (__bf16)ra[i].z;
            As[buf][rA[i]][cA + 3] = (__bf16)ra[i].w;
            Bt[buf][cA + 0][rA[i]] = (__bf16)rb[i].x;    // transpose
            Bt[buf][cA + 1][rA[i]] = (__bf16)rb[i].y;
            Bt[buf][cA + 2][rA[i]] = (__bf16)rb[i].z;
            Bt[buf][cA + 3][rA[i]] = (__bf16)rb[i].w;
        }
    };

    // ---- 8 WMMAs on one LDS buffer ---------------------------------------
    auto compute = [&](int buf) {
        #pragma unroll
        for (int ks = 0; ks < BKT; ks += 32) {
            // A fragment (ISA 16-bit A 16x32): elems 0..7 = K[ks+hi*8 ..],
            // elems 8..15 = K[ks+16+hi*8 ..]  -> two ds_load_b128
            const int am = wave * 16 + lo;
            bf16x8 a0 = *(const bf16x8*)&As[buf][am][ks + hi * 8];
            bf16x8 a1 = *(const bf16x8*)&As[buf][am][ks + 16 + hi * 8];
            bf16x16 af = __builtin_shufflevector(a0, a1,
                0,1,2,3,4,5,6,7,8,9,10,11,12,13,14,15);
            #pragma unroll
            for (int tn = 0; tn < 4; ++tn) {
                // B fragment (16-bit B 32x16): elems 0..7 = K[ks+hi*8 ..],
                // elems 8..15 = K[ks+16+hi*8 ..] for column N=lo
                const int bn = tn * 16 + lo;
                bf16x8 b0 = *(const bf16x8*)&Bt[buf][bn][ks + hi * 8];
                bf16x8 b1 = *(const bf16x8*)&Bt[buf][bn][ks + 16 + hi * 8];
                bf16x16 bf = __builtin_shufflevector(b0, b1,
                    0,1,2,3,4,5,6,7,8,9,10,11,12,13,14,15);
                acc[tn] = __builtin_amdgcn_wmma_f32_16x16x32_bf16(
                    false, af, false, bf, (short)0, acc[tn], false, false);
            }
        }
    };

    const int ntiles = K / BKT;

    // prologue: tile 0
    loadRegs(0);
    storeLDS(0);
    __syncthreads();

    for (int kt = 0; kt < ntiles; ++kt) {
        const int  cur      = kt & 1;
        const bool haveNext = (kt + 1 < ntiles);
        if (haveNext) {
            loadRegs((kt + 1) * BKT);          // issue loads, no wait yet
            if (kt + 2 < ntiles) {             // pull tile kt+2 into L2
                __builtin_prefetch(A + (size_t)(m0 + rA[0]) * K + (kt + 2) * BKT + cA, 0, 1);
                int pn = n0 + cA; if (pn >= N) pn = N - 1;
                __builtin_prefetch(B + (size_t)((kt + 2) * BKT + rA[0]) * N + pn, 0, 1);
            }
        }
        compute(cur);                           // WMMAs hide load latency
        if (haveNext) storeLDS(cur ^ 1);        // cvt + ds_store after compute
        __syncthreads();
    }

    // ---- write-back: VGPR r of a 16x16 f32 D holds M = r + hi*8, N = lo
    const int mrow = m0 + wave * 16 + hi * 8;
    #pragma unroll
    for (int tn = 0; tn < 4; ++tn) {
        int col = n0 + tn * 16 + lo;
        if (col < N) {
            float bv = bias ? bias[col] : 0.0f;
            #pragma unroll
            for (int r = 0; r < 8; ++r) {
                float v = acc[tn][r] + bv;
                if (relu) v = fmaxf(v, 0.0f);
                C[(size_t)(mrow + r) * N + col] = v;
            }
        }
    }
}

// -------------------- causal depthwise conv + silu -------------------------
__global__ __launch_bounds__(256) void conv_silu_k(
    const float* __restrict__ zx, const float* __restrict__ cw,
    const float* __restrict__ cb, float* __restrict__ out)
{
    int idx = blockIdx.x * 256 + threadIdx.x;   // ROWS * CONV_DIM
    if (idx >= ROWS * CONV_DIM) return;
    int row = idx / CONV_DIM;
    int c   = idx - row * CONV_DIM;
    int b   = row >> 3, l = row & 7;
    float acc = cb[c];
    #pragma unroll
    for (int k = 0; k < DCONV; ++k) {
        int ls = l + k - (DCONV - 1);
        if (ls >= 0)
            acc += zx[(size_t)(b * SEQL + ls) * DPROJ + D_INNER + c] * cw[c * DCONV + k];
    }
    out[idx] = acc * (1.0f / (1.0f + __expf(-acc)));   // silu
}

// -------------------- dt = softplus(dt+bias); dA = exp(-exp(A_log)*dt) -----
__global__ __launch_bounds__(256) void dt_da_k(
    const float* __restrict__ zx, const float* __restrict__ dt_bias,
    const float* __restrict__ A_log, float* __restrict__ dtb,
    float* __restrict__ dab)
{
    int idx = blockIdx.x * 256 + threadIdx.x;   // ROWS * NHEADS
    if (idx >= ROWS * NHEADS) return;
    int row = idx >> 5, h = idx & 31;
    float v  = zx[(size_t)row * DPROJ + (D_INNER + CONV_DIM) + h] + dt_bias[h];
    float sp = (v > 20.0f) ? v : log1pf(__expf(v));
    dtb[idx] = sp;
    dab[idx] = __expf(-__expf(A_log[h]) * sp);
}

// -------------------- SSM scan: one block per (batch, head) ----------------
// State 64x128 fp32 lives in registers: thread t owns p = t>>2, s-slice of 32.
__global__ __launch_bounds__(256) void mamba_scan_k(
    const float* __restrict__ xbc, const float* __restrict__ dtb,
    const float* __restrict__ dab, float* __restrict__ ybuf)
{
    const int b  = blockIdx.x >> 5;
    const int h  = blockIdx.x & 31;
    const int t  = threadIdx.x;
    const int p  = t >> 2;
    const int sb = t & 3;
    __shared__ float Bs[DSTATE];
    __shared__ float Cs[DSTATE];

    float state[32];
    #pragma unroll
    for (int i = 0; i < 32; ++i) state[i] = 0.0f;

    for (int l = 0; l < SEQL; ++l) {
        const int row = b * SEQL + l;
        const float* base = xbc + (size_t)row * CONV_DIM;
        if (t < DSTATE)            Bs[t]          = base[D_INNER + t];
        else if (t < 2 * DSTATE)   Cs[t - DSTATE] = base[D_INNER + DSTATE + (t - DSTATE)];
        __syncthreads();

        const float xv   = base[h * HEADDIM + p];
        const float dA   = dab[row * NHEADS + h];
        const float coef = dtb[row * NHEADS + h] * xv;
        const int   s0   = sb * 32;
        float ysum = 0.0f;
        #pragma unroll
        for (int i = 0; i < 32; ++i) {
            float st = dA * state[i] + coef * Bs[s0 + i];
            state[i] = st;
            ysum += st * Cs[s0 + i];
        }
        // reduce over the 4 lanes sharing one p (same wave32)
        ysum += __shfl_xor(ysum, 1, 32);
        ysum += __shfl_xor(ysum, 2, 32);
        if (sb == 0)
            ybuf[(size_t)row * D_INNER + h * HEADDIM + p] = ysum;
        __syncthreads();
    }
}

// -------------------- y = (y + xs*D) * silu(z); RMSNorm * w ----------------
__global__ __launch_bounds__(256) void gate_rmsnorm_k(
    const float* __restrict__ ybuf, const float* __restrict__ xbc,
    const float* __restrict__ zx, const float* __restrict__ Dp,
    const float* __restrict__ nw, float* __restrict__ out)
{
    const int row = blockIdx.x;
    const int t   = threadIdx.x;
    __shared__ float red[256];
    float vals[D_INNER / 256];
    float ss = 0.0f;
    #pragma unroll
    for (int j = 0; j < D_INNER / 256; ++j) {
        int c   = t + j * 256;
        float y = ybuf[(size_t)row * D_INNER + c]
                + xbc[(size_t)row * CONV_DIM + c] * Dp[c];
        float z = zx[(size_t)row * DPROJ + c];
        y *= z * (1.0f / (1.0f + __expf(-z)));
        vals[j] = y;
        ss += y * y;
    }
    red[t] = ss;
    __syncthreads();
    for (int ofs = 128; ofs > 0; ofs >>= 1) {
        if (t < ofs) red[t] += red[t + ofs];
        __syncthreads();
    }
    const float scale = rsqrtf(red[0] * (1.0f / D_INNER) + 1e-5f);
    #pragma unroll
    for (int j = 0; j < D_INNER / 256; ++j) {
        int c = t + j * 256;
        out[(size_t)row * D_INNER + c] = vals[j] * scale * nw[c];
    }
}

// -------------------- final tiny head: [256x1024] @ [1024x2] + b -----------
__global__ __launch_bounds__(256) void head_k(
    const float* __restrict__ f1, const float* __restrict__ w2,
    const float* __restrict__ b2, float* __restrict__ out)
{
    const int b = blockIdx.x;
    const int t = threadIdx.x;
    __shared__ float red0[256];
    __shared__ float red1[256];
    float a0 = 0.0f, a1 = 0.0f;
    for (int k = t; k < HID; k += 256) {
        float v = f1[(size_t)b * HID + k];
        a0 += v * w2[k * 2 + 0];
        a1 += v * w2[k * 2 + 1];
    }
    red0[t] = a0; red1[t] = a1;
    __syncthreads();
    for (int ofs = 128; ofs > 0; ofs >>= 1) {
        if (t < ofs) { red0[t] += red0[t + ofs]; red1[t] += red1[t + ofs]; }
        __syncthreads();
    }
    if (t == 0) {
        out[b * 2 + 0] = red0[0] + b2[0];
        out[b * 2 + 1] = red1[0] + b2[1];
    }
}

// ---------------------------------------------------------------------------
extern "C" void kernel_launch(void* const* d_in, const int* in_sizes, int n_in,
                              void* d_out, int out_size, void* d_ws, size_t ws_size,
                              hipStream_t stream)
{
    const float* x      = (const float*)d_in[0];
    const float* w_in1  = (const float*)d_in[1];
    const float* b_in1  = (const float*)d_in[2];
    const float* w_in2  = (const float*)d_in[3];
    const float* b_in2  = (const float*)d_in[4];
    const float* w_out1 = (const float*)d_in[21];
    const float* b_out1 = (const float*)d_in[22];
    const float* w_out2 = (const float*)d_in[23];
    const float* b_out2 = (const float*)d_in[24];

    float* ws = (float*)d_ws;
    size_t off = 0;
    float* h1    = ws + off; off += (size_t)BATCH * HID * SEQL;   // 256x8192
    float* h2    = ws + off; off += (size_t)BATCH * HID * SEQL;   // 256x8192
    float* hbuf  = ws + off; off += (size_t)ROWS * HID;           // 2048x1024
    float* zx    = ws + off; off += (size_t)ROWS * DPROJ;         // 2048x4384
    float* xbc   = ws + off; off += (size_t)ROWS * CONV_DIM;      // 2048x2304
    float* dtb   = ws + off; off += (size_t)ROWS * NHEADS;
    float* dab   = ws + off; off += (size_t)ROWS * NHEADS;
    float* ybuf  = ws + off; off += (size_t)ROWS * D_INNER;       // 2048x2048
    float* ynorm = ws + off; off += (size_t)ROWS * D_INNER;       // 2048x2048
    float* f1    = ws + off; off += (size_t)BATCH * HID;          // 256x1024

    // ---- input MLP ----
    gemm_bf16_wmma<<<dim3((HID * SEQL) / BN, BATCH / BM), 128, 0, stream>>>(
        x, w_in1, b_in1, h1, BATCH, HID * SEQL, HID, 1);
    gemm_bf16_wmma<<<dim3((HID * SEQL) / BN, BATCH / BM), 128, 0, stream>>>(
        h1, w_in2, b_in2, h2, BATCH, HID * SEQL, HID * SEQL, 0);

    // ---- one Mamba2 block ----
    auto mamba = [&](const float* hin, float* hout, int pbase) {
        const float* in_proj  = (const float*)d_in[pbase + 0];
        const float* conv_w   = (const float*)d_in[pbase + 1];
        const float* conv_b   = (const float*)d_in[pbase + 2];
        const float* dt_bias  = (const float*)d_in[pbase + 3];
        const float* A_log    = (const float*)d_in[pbase + 4];
        const float* Dp       = (const float*)d_in[pbase + 5];
        const float* norm_w   = (const float*)d_in[pbase + 6];
        const float* out_proj = (const float*)d_in[pbase + 7];

        gemm_bf16_wmma<<<dim3((DPROJ + BN - 1) / BN, ROWS / BM), 128, 0, stream>>>(
            hin, in_proj, nullptr, zx, ROWS, DPROJ, HID, 0);
        conv_silu_k<<<(ROWS * CONV_DIM + 255) / 256, 256, 0, stream>>>(
            zx, conv_w, conv_b, xbc);
        dt_da_k<<<(ROWS * NHEADS + 255) / 256, 256, 0, stream>>>(
            zx, dt_bias, A_log, dtb, dab);
        mamba_scan_k<<<BATCH * NHEADS, 256, 0, stream>>>(xbc, dtb, dab, ybuf);
        gate_rmsnorm_k<<<ROWS, 256, 0, stream>>>(ybuf, xbc, zx, Dp, norm_w, ynorm);
        gemm_bf16_wmma<<<dim3(HID / BN, ROWS / BM), 128, 0, stream>>>(
            ynorm, out_proj, nullptr, hout, ROWS, HID, D_INNER, 0);
    };

    mamba(h2, hbuf, 5);    // layer 1: h2 -> hbuf
    mamba(hbuf, h2, 13);   // layer 2: hbuf -> h2 (reuse)

    // ---- output MLP ----
    gemm_bf16_wmma<<<dim3(HID / BN, BATCH / BM), 128, 0, stream>>>(
        h2, w_out1, b_out1, f1, BATCH, HID, HID * SEQL, 1);
    head_k<<<BATCH, 256, 0, stream>>>(f1, w_out2, b_out2, (float*)d_out);
}
